// CRLossColBERT_74466142978313
// MI455X (gfx1250) — compile-verified, matrix operand order checked by reading, not run on hardware
//
#include <hip/hip_runtime.h>
#include <hip/hip_bf16.h>
#include <math.h>

// ---------------------------------------------------------------------------
// ColBERT max-sim + weighted RankNet loss for MI455X (gfx1250, wave32, WMMA)
//   B=64, N=16, Lq=32, Ld=256, D=128
// ---------------------------------------------------------------------------

typedef __attribute__((ext_vector_type(16))) _Float16 v16h;
typedef __attribute__((ext_vector_type(8)))  _Float16 v8h;
typedef __attribute__((ext_vector_type(4)))  _Float16 v4h;
typedef __attribute__((ext_vector_type(8)))  float    v8f;

#define BATCH   64
#define NDOCS   16
#define LQ      32
#define LD      256
#define DIM     128
#define HALF_LD 128                  // doc rows staged per LDS pass
#define ROWPAD  136                  // 128 + 8 halves -> 272B row stride (bank skew)

// ---------------------------------------------------------------------------
// Kernel 1: L2-normalize queries, emit f16 row-major [B*Lq][128]
// grid = 64 blocks (one per batch), 256 threads; 8 lanes cooperate per row.
// ---------------------------------------------------------------------------
__global__ __launch_bounds__(256) void qnorm_kernel(const float* __restrict__ q,
                                                    _Float16* __restrict__ qn) {
    const int tid    = threadIdx.x;
    const int laneIn = tid & 7;          // lane within 8-lane row group
    const int rowLoc = tid >> 3;         // 0..31  (Lq rows)
    const size_t row = (size_t)blockIdx.x * LQ + rowLoc;

    const float4* src = (const float4*)(q + row * DIM);
    float v[16];
    float ss = 0.0f;
#pragma unroll
    for (int i = 0; i < 4; ++i) {
        float4 x = src[laneIn + 8 * i];
        v[4*i+0] = x.x; v[4*i+1] = x.y; v[4*i+2] = x.z; v[4*i+3] = x.w;
        ss += x.x*x.x + x.y*x.y + x.z*x.z + x.w*x.w;
    }
#pragma unroll
    for (int m = 1; m <= 4; m <<= 1) ss += __shfl_xor(ss, m, 32);
    const float inv = 1.0f / fmaxf(sqrtf(ss), 1e-12f);

    _Float16* dst = qn + row * DIM;
#pragma unroll
    for (int i = 0; i < 4; ++i) {
        v4h h;
        h.x = (_Float16)(v[4*i+0] * inv);
        h.y = (_Float16)(v[4*i+1] * inv);
        h.z = (_Float16)(v[4*i+2] * inv);
        h.w = (_Float16)(v[4*i+3] * inv);
        *(v4h*)(dst + (laneIn + 8 * i) * 4) = h;
    }
}

// ---------------------------------------------------------------------------
// Kernel 2: per-(b,n) max-sim score via v_wmma_f32_16x16x32_f16
// grid = 1024 blocks (b*16+n), 256 threads = 8 waves.
//   wave w: mtile = w&1 (query rows 16*mtile..), ngrp = w>>1 (doc col group)
// Doc tile streamed in two 128-row halves through LDS (mask+normalize+f16).
// ---------------------------------------------------------------------------
__global__ __launch_bounds__(256) void score_kernel(const float* __restrict__ doc,
                                                    const int*   __restrict__ dmask,
                                                    const _Float16* __restrict__ qn,
                                                    float* __restrict__ scores) {
    __shared__ _Float16 sD[HALF_LD][ROWPAD];     // 34,816 B
    __shared__ float    pmax[2][16][4];          // [mtile][row][ngrp]

    const int bn   = blockIdx.x;                 // b*16 + n
    const int b    = bn >> 4;
    const int tid  = threadIdx.x;
    const int lane = tid & 31;
    const int wave = tid >> 5;
    const int mtile = wave & 1;
    const int ngrp  = wave >> 1;                 // 0..3
    const int lrow  = lane & 15;                 // row/col within 16-tile
    const int g     = lane >> 4;                 // lane half -> k-chunk select

    // --- cache A fragments (query) for all 4 k-steps, ISA 16-bit A layout ---
    const _Float16* qbase = qn + ((size_t)b * LQ + mtile * 16 + lrow) * DIM;
    v16h afrag[4];
#pragma unroll
    for (int ks = 0; ks < 4; ++ks) {
        const int kb = ks * 32;
        v8h lo = *(const v8h*)(qbase + kb + g * 8);
        v8h hi = *(const v8h*)(qbase + kb + 16 + g * 8);
#pragma unroll
        for (int i = 0; i < 8; ++i) { afrag[ks][i] = lo[i]; afrag[ks][i + 8] = hi[i]; }
    }

    float rowmax[8];
#pragma unroll
    for (int r = 0; r < 8; ++r) rowmax[r] = -3.0e38f;

    const float* docBase  = doc + (size_t)bn * LD * DIM;
    const int*   maskBase = dmask + (size_t)bn * LD;

    for (int half = 0; half < 2; ++half) {
        // ---- fill: mask, f32 normalize, f16 convert into LDS ----
        const int laneIn = tid & 7;
        const int grp    = tid >> 3;             // 0..31 row groups
#pragma unroll
        for (int it = 0; it < 4; ++it) {
            const int r  = grp + 32 * it;        // local row 0..127
            const int gr = half * HALF_LD + r;   // global doc row
            const float m = (float)maskBase[gr];
            const float4* src = (const float4*)(docBase + (size_t)gr * DIM);
            float v[16];
            float ss = 0.0f;
#pragma unroll
            for (int i = 0; i < 4; ++i) {
                float4 x = src[laneIn + 8 * i];
                x.x *= m; x.y *= m; x.z *= m; x.w *= m;
                v[4*i+0] = x.x; v[4*i+1] = x.y; v[4*i+2] = x.z; v[4*i+3] = x.w;
                ss += x.x*x.x + x.y*x.y + x.z*x.z + x.w*x.w;
            }
#pragma unroll
            for (int s = 1; s <= 4; s <<= 1) ss += __shfl_xor(ss, s, 32);
            const float inv = 1.0f / fmaxf(sqrtf(ss), 1e-12f);
#pragma unroll
            for (int i = 0; i < 4; ++i) {
                v4h h;
                h.x = (_Float16)(v[4*i+0] * inv);
                h.y = (_Float16)(v[4*i+1] * inv);
                h.z = (_Float16)(v[4*i+2] * inv);
                h.w = (_Float16)(v[4*i+3] * inv);
                *(v4h*)(&sD[r][(laneIn + 8 * i) * 4]) = h;
            }
        }
        __syncthreads();

        // ---- GEMM: each wave does 2 local n-tiles x 4 k-steps ----
#pragma unroll
        for (int t = 0; t < 2; ++t) {
            const int lt = 2 * ngrp + t;                 // local n-tile 0..7
            const _Float16* drow = &sD[lt * 16 + lrow][0];
            v8f acc = {};
#pragma unroll
            for (int ks = 0; ks < 4; ++ks) {
                const int kb = ks * 32;
                v8h lo = *(const v8h*)(drow + kb + g * 8);
                v8h hi = *(const v8h*)(drow + kb + 16 + g * 8);
                v16h bfrag;
#pragma unroll
                for (int i = 0; i < 8; ++i) { bfrag[i] = lo[i]; bfrag[i + 8] = hi[i]; }
                acc = __builtin_amdgcn_wmma_f32_16x16x32_f16(
                        false, afrag[ks], false, bfrag, (short)0, acc, false, false);
            }
#pragma unroll
            for (int r = 0; r < 8; ++r) rowmax[r] = fmaxf(rowmax[r], acc[r]);
        }
        __syncthreads();   // before next half overwrites sD
    }

    // ---- reduce: max over n (16 lanes per half), combine n-groups, sum rows ----
#pragma unroll
    for (int r = 0; r < 8; ++r) {
#pragma unroll
        for (int s = 1; s <= 8; s <<= 1)
            rowmax[r] = fmaxf(rowmax[r], __shfl_xor(rowmax[r], s, 32));
    }
    if (lrow == 0) {
#pragma unroll
        for (int r = 0; r < 8; ++r) pmax[mtile][g * 8 + r][ngrp] = rowmax[r];
    }
    __syncthreads();

    if (tid < 32) {
        const int mt = tid >> 4, rw = tid & 15;
        float v = pmax[mt][rw][0];
        v = fmaxf(v, pmax[mt][rw][1]);
        v = fmaxf(v, pmax[mt][rw][2]);
        v = fmaxf(v, pmax[mt][rw][3]);
#pragma unroll
        for (int s = 16; s >= 1; s >>= 1) v += __shfl_xor(v, s, 32);
        if (tid == 0) scores[bn] = v;
    }
}

// ---------------------------------------------------------------------------
// Kernel 3: weighted pairwise RankNet loss per batch (faithful triu semantics)
// grid = 64 blocks, 256 threads (one pair each).
// ---------------------------------------------------------------------------
__global__ __launch_bounds__(256) void loss_kernel(const float* __restrict__ scores,
                                                   const float* __restrict__ labels,
                                                   float* __restrict__ partial) {
    __shared__ float sS[NDOCS], sY[NDOCS], sI[NDOCS];
    __shared__ float wsum[8];
    const int b   = blockIdx.x;
    const int tid = threadIdx.x;
    if (tid < NDOCS) {
        sS[tid] = scores[b * NDOCS + tid];
        sY[tid] = labels[b * 2 * NDOCS + tid];
        sI[tid] = labels[b * 2 * NDOCS + NDOCS + tid];
    }
    __syncthreads();

    const int i = tid >> 4, j = tid & 15;
    const float diff = (i <= j) ? (sS[i] - sS[j]) : 0.0f;   // jnp.triu on diffs
    const float w    = fabsf(sI[i] - sI[j]);
    float acc = (sY[i] > sY[j]) ? log1pf(expf(-diff)) * w : 0.0f;

#pragma unroll
    for (int s = 16; s >= 1; s >>= 1) acc += __shfl_xor(acc, s, 32);
    if ((tid & 31) == 0) wsum[tid >> 5] = acc;
    __syncthreads();
    if (tid == 0) {
        float t = 0.0f;
#pragma unroll
        for (int k = 0; k < 8; ++k) t += wsum[k];
        partial[b] = t;
    }
}

__global__ void reduce_kernel(const float* __restrict__ partial, float* __restrict__ out) {
    float s = 0.0f;
#pragma unroll
    for (int k = 0; k < BATCH; ++k) s += partial[k];
    out[0] = s;
}

// ---------------------------------------------------------------------------
extern "C" void kernel_launch(void* const* d_in, const int* in_sizes, int n_in,
                              void* d_out, int out_size, void* d_ws, size_t ws_size,
                              hipStream_t stream) {
    const float* query  = (const float*)d_in[0];   // [64,32,128]
    const float* doc    = (const float*)d_in[1];   // [64,16,256,128]
    const int*   dmask  = (const int*)  d_in[2];   // [64,16,256]
    const float* labels = (const float*)d_in[3];   // [64,32]
    float* out = (float*)d_out;

    char* ws = (char*)d_ws;
    _Float16* qn   = (_Float16*)(ws);                       // 512 KB
    float* scores  = (float*)(ws + (size_t)BATCH*LQ*DIM*2); //   4 KB
    float* partial = scores + BATCH * NDOCS;                //  256 B

    qnorm_kernel <<<BATCH,           256, 0, stream>>>(query, qn);
    score_kernel <<<BATCH * NDOCS,   256, 0, stream>>>(doc, dmask, qn, scores);
    loss_kernel  <<<BATCH,           256, 0, stream>>>(scores, labels, partial);
    reduce_kernel<<<1,               1,   0, stream>>>(partial, out);
}
